// UniCMP_70291434766743
// MI455X (gfx1250) — matrix operand order repaired and checked by reading, not compiled, not run on hardware
//
#include <hip/hip_runtime.h>
#include <cstdint>
#include <cstddef>

#define N_NODES 50000
#define E_EDGES 150000
#define R_REL   3
#define L_LAYERS 2
#define HID     512
#define HEADS   8
#define DHD     64
#define NCLS    23

typedef __attribute__((ext_vector_type(16))) __bf16 v16bf;
typedef __attribute__((ext_vector_type(8)))  float  v8f;

union BF16x8 { uint4 u; __bf16 h[8]; };
union AFrag  { v16bf v; __bf16 h[16]; };

__device__ __forceinline__ unsigned short f32_to_bf16_rte(float f) {
    unsigned u = __float_as_uint(f);
    unsigned r = u + 0x7FFFu + ((u >> 16) & 1u);
    return (unsigned short)(r >> 16);
}

__device__ __forceinline__ unsigned enc_ord(float f) {
    unsigned u = __float_as_uint(f);
    return (u >> 31) ? ~u : (u | 0x80000000u);
}
__device__ __forceinline__ float dec_ord(unsigned u) {
    return (u & 0x80000000u) ? __uint_as_float(u & 0x7FFFFFFFu) : __uint_as_float(~u);
}

// ---------------- f32 -> bf16 conversion ----------------
__global__ void f32_to_bf16_kernel(const float* __restrict__ s,
                                   unsigned short* __restrict__ d, size_t n) {
    size_t i = (size_t)blockIdx.x * blockDim.x + threadIdx.x;
    if (i < n) d[i] = f32_to_bf16_rte(s[i]);
}

// ---------------- WMMA GEMM: C[M,512] = A[M,512](bf16) x B[512,512](bf16) (+bias) ----------
// Block: 256 threads = 8 waves. Block tile 256(M) x 64(N).
// Whole 512x64 B panel staged transposed in LDS once (64KB of the WGP's 320KB);
// k-loop then runs barrier-free: 4 global_load_b128 (A) + 8 ds_load_b128 (B) + 8 WMMA.
__global__ __launch_bounds__(256)
void gemm_bf16_wmma(const unsigned short* __restrict__ Au,
                    const unsigned short* __restrict__ Bu,
                    const float* __restrict__ bias,
                    float* __restrict__ C, int M) {
    const int K = HID, Nc = HID;
    __shared__ __bf16 Bt[64 * HID];   // Bt[col][k], 64 cols x 512 k, 64KB

    const __bf16* A = (const __bf16*)Au;
    const __bf16* B = (const __bf16*)Bu;

    int tid  = threadIdx.x;
    int wave = tid >> 5;
    int lane = tid & 31;
    int half = lane >> 4;     // 0: lanes 0-15, 1: lanes 16-31
    int mr   = lane & 15;

    int m0  = blockIdx.x * 256 + wave * 32;
    int bn0 = blockIdx.y * 64;

    // stage full B panel transposed into LDS (once per block)
#pragma unroll
    for (int i = 0; i < 16; ++i) {
        int c   = tid + i * 256;      // 0..4095 chunk id
        int row = c >> 3;             // K index 0..511
        int cg  = (c & 7) * 8;        // col group 0..56
        BF16x8 bv = *(const BF16x8*)(B + (size_t)row * Nc + bn0 + cg);
#pragma unroll
        for (int j = 0; j < 8; ++j) Bt[(cg + j) * HID + row] = bv.h[j];
    }
    __syncthreads();

    // clamp rows (EXEC must stay all-ones for WMMA); stores are predicated below
    int r0 = m0 + mr;      if (r0 >= M) r0 = M - 1;
    int r1 = m0 + 16 + mr; if (r1 >= M) r1 = M - 1;
    const __bf16* A0 = A + (size_t)r0 * K + half * 8;
    const __bf16* A1 = A + (size_t)r1 * K + half * 8;

    v8f acc[2][4] = {};

    for (int k0 = 0; k0 < K; k0 += 32) {
        // A fragments 16x32: lanes 0-15 K=0..7 & 16..23, lanes 16-31 K=8..15 & 24..31
        AFrag a0, a1;
        BF16x8 x0 = *(const BF16x8*)(A0 + k0);
        BF16x8 x1 = *(const BF16x8*)(A0 + k0 + 16);
        BF16x8 y0 = *(const BF16x8*)(A1 + k0);
        BF16x8 y1 = *(const BF16x8*)(A1 + k0 + 16);
#pragma unroll
        for (int j = 0; j < 8; ++j) {
            a0.h[j] = x0.h[j]; a0.h[j + 8] = x1.h[j];
            a1.h[j] = y0.h[j]; a1.h[j + 8] = y1.h[j];
        }
#pragma unroll
        for (int t = 0; t < 4; ++t) {
            // B fragment 32x16: col = lane&15, lanes 0-15 K=0..7 & 16..23, lanes 16-31 K=8..15 & 24..31
            AFrag b;
            const __bf16* bp = &Bt[(t * 16 + mr) * HID + k0 + half * 8];
            BF16x8 b0 = *(const BF16x8*)(bp);
            BF16x8 b1 = *(const BF16x8*)(bp + 16);
#pragma unroll
            for (int j = 0; j < 8; ++j) { b.h[j] = b0.h[j]; b.h[j + 8] = b1.h[j]; }
            acc[0][t] = __builtin_amdgcn_wmma_f32_16x16x32_bf16(
                false, a0.v, false, b.v, (short)0, acc[0][t], false, false);
            acc[1][t] = __builtin_amdgcn_wmma_f32_16x16x32_bf16(
                false, a1.v, false, b.v, (short)0, acc[1][t], false, false);
        }
    }

#pragma unroll
    for (int mi = 0; mi < 2; ++mi)
#pragma unroll
        for (int t = 0; t < 4; ++t) {
            int c = bn0 + t * 16 + mr;
            float bi = bias ? bias[c] : 0.0f;
#pragma unroll
            for (int rr = 0; rr < 8; ++rr) {
                int row = m0 + mi * 16 + rr + half * 8;
                if (row < M) C[(size_t)row * Nc + c] = acc[mi][t][rr] + bi;
            }
        }
}

// ---------------- degrees ----------------
__global__ void degree_kernel(const int* __restrict__ src, const int* __restrict__ dst,
                              float* degO, float* degI) {
    int e = blockIdx.x * blockDim.x + threadIdx.x;
    if (e >= E_EDGES) return;
    atomicAdd(&degO[src[e]], 1.0f);
    atomicAdd(&degI[dst[e]], 1.0f);
}

// x[n,f] *= rsqrt(max(deg[n],1))
__global__ void rowscale_kernel(float* __restrict__ x, const float* __restrict__ deg) {
    size_t i = (size_t)blockIdx.x * blockDim.x + threadIdx.x;
    if (i >= (size_t)N_NODES * HID) return;
    int n = (int)(i / HID);
    x[i] *= rsqrtf(fmaxf(deg[n], 1.0f));
}

// out[dst] += x[src]   (one 128-thread block per edge, float4 per thread)
__global__ __launch_bounds__(128)
void scatter_sum_kernel(const float* __restrict__ x, const int* __restrict__ src,
                        const int* __restrict__ dst, float* __restrict__ out) {
    int e = blockIdx.x;
    int s = src[e], d = dst[e];
    int f = threadIdx.x * 4;
    const float4 v = *(const float4*)(x + (size_t)s * HID + f);
    float* o = out + (size_t)d * HID + f;
    atomicAdd(o + 0, v.x); atomicAdd(o + 1, v.y);
    atomicAdd(o + 2, v.z); atomicAdd(o + 3, v.w);
}

// fused: y = in*rowscale + bias -> layernorm(g,b) -> elu -> store/add into S[n,r,:]
__global__ __launch_bounds__(256)
void ln_elu_kernel(const float* __restrict__ in, const float* __restrict__ rowdeg,
                   const float* __restrict__ bias, const float* __restrict__ gamma,
                   const float* __restrict__ beta, float* __restrict__ S,
                   int r, int addmode) {
    __shared__ float red[18];
    int n = blockIdx.x;
    int tid = threadIdx.x;
    int f0 = tid * 2;
    float rs = rowdeg ? rsqrtf(fmaxf(rowdeg[n], 1.0f)) : 1.0f;
    const float* row = in + (size_t)n * HID;
    float v0 = row[f0]     * rs + (bias ? bias[f0]     : 0.0f);
    float v1 = row[f0 + 1] * rs + (bias ? bias[f0 + 1] : 0.0f);
    float s  = v0 + v1;
    float ss = v0 * v0 + v1 * v1;
#pragma unroll
    for (int off = 16; off; off >>= 1) {
        s  += __shfl_xor(s, off, 32);
        ss += __shfl_xor(ss, off, 32);
    }
    int wave = tid >> 5, lane = tid & 31;
    if (lane == 0) { red[wave] = s; red[8 + wave] = ss; }
    __syncthreads();
    if (tid == 0) {
        float ts = 0.f, tss = 0.f;
        for (int i = 0; i < 8; ++i) { ts += red[i]; tss += red[8 + i]; }
        red[16] = ts * (1.0f / HID);
        red[17] = tss * (1.0f / HID);
    }
    __syncthreads();
    float u   = red[16];
    float var = red[17] - u * u;
    float inv = rsqrtf(var + 1e-12f);
    float o0 = gamma[f0]     * (v0 - u) * inv + beta[f0];
    float o1 = gamma[f0 + 1] * (v1 - u) * inv + beta[f0 + 1];
    o0 = o0 > 0.f ? o0 : (expf(o0) - 1.f);
    o1 = o1 > 0.f ? o1 : (expf(o1) - 1.f);
    float* so = S + ((size_t)n * R_REL + r) * HID + f0;
    if (addmode) { so[0] += o0; so[1] += o1; }
    else         { so[0]  = o0; so[1]  = o1; }
}

// el/er per (node, head)
__global__ void gat_elr_kernel(const float* __restrict__ f, const float* __restrict__ al,
                               const float* __restrict__ ar, float* __restrict__ el,
                               float* __restrict__ er) {
    int i = blockIdx.x * blockDim.x + threadIdx.x;
    if (i >= N_NODES * HEADS) return;
    int n = i / HEADS, h = i % HEADS;
    const float* fr = f + (size_t)n * HID + h * DHD;
    const float* a1 = al + h * DHD;
    const float* a2 = ar + h * DHD;
    float s1 = 0.f, s2 = 0.f;
    for (int d = 0; d < DHD; ++d) { float x = fr[d]; s1 += x * a1[d]; s2 += x * a2[d]; }
    el[i] = s1; er[i] = s2;
}

__global__ void edge_logit_kernel(const int* __restrict__ src, const int* __restrict__ dst,
                                  const float* __restrict__ el, const float* __restrict__ er,
                                  float* __restrict__ ebuf, unsigned* __restrict__ menc) {
    int i = blockIdx.x * blockDim.x + threadIdx.x;
    if (i >= E_EDGES * HEADS) return;
    int e = i / HEADS, h = i % HEADS;
    float x = el[src[e] * HEADS + h] + er[dst[e] * HEADS + h];
    x = x > 0.f ? x : 0.2f * x;                 // leaky relu
    ebuf[i] = x;
    atomicMax(&menc[dst[e] * HEADS + h], enc_ord(x));
}

__global__ void edge_expsum_kernel(const int* __restrict__ dst, float* __restrict__ ebuf,
                                   const unsigned* __restrict__ menc, float* __restrict__ ssum) {
    int i = blockIdx.x * blockDim.x + threadIdx.x;
    if (i >= E_EDGES * HEADS) return;
    int e = i / HEADS, h = i % HEADS;
    int idx = dst[e] * HEADS + h;
    float ex = expf(ebuf[i] - dec_ord(menc[idx]));
    ebuf[i] = ex;
    atomicAdd(&ssum[idx], ex);
}

__global__ __launch_bounds__(128)
void gat_scatter_kernel(const float* __restrict__ f, const int* __restrict__ src,
                        const int* __restrict__ dst, const float* __restrict__ ebuf,
                        const float* __restrict__ ssum, float* __restrict__ out) {
    int e = blockIdx.x;
    int s = src[e], d = dst[e];
    int f0 = threadIdx.x * 4;
    int h  = f0 >> 6;
    float sum = ssum[d * HEADS + h];
    float alpha = ebuf[e * HEADS + h] / (sum > 0.f ? sum : 1.f);
    const float4 v = *(const float4*)(f + (size_t)s * HID + f0);
    float* o = out + (size_t)d * HID + f0;
    atomicAdd(o + 0, alpha * v.x); atomicAdd(o + 1, alpha * v.y);
    atomicAdd(o + 2, alpha * v.z); atomicAdd(o + 3, alpha * v.w);
}

// relation MHA over R=3 tokens; block = node, wave = head, 2 dims/lane
__global__ __launch_bounds__(256)
void mha_kernel(const float* __restrict__ q, const float* __restrict__ k,
                const float* __restrict__ v, float* __restrict__ out) {
    int n    = blockIdx.x;
    int head = threadIdx.x >> 5;
    int lane = threadIdx.x & 31;
    int d0   = lane * 2;
    size_t base = (size_t)n * R_REL * HID + head * DHD + d0;
    float qr[3][2], kr[3][2], vr[3][2];
#pragma unroll
    for (int r = 0; r < 3; ++r) {
        size_t idx = base + (size_t)r * HID;
        qr[r][0] = q[idx]; qr[r][1] = q[idx + 1];
        kr[r][0] = k[idx]; kr[r][1] = k[idx + 1];
        vr[r][0] = v[idx]; vr[r][1] = v[idx + 1];
    }
    float p[3][3];
#pragma unroll
    for (int r = 0; r < 3; ++r)
#pragma unroll
        for (int s = 0; s < 3; ++s) {
            float t = qr[r][0] * kr[s][0] + qr[r][1] * kr[s][1];
#pragma unroll
            for (int off = 16; off; off >>= 1) t += __shfl_xor(t, off, 32);
            p[r][s] = t * 0.125f;   // 1/sqrt(64)
        }
    float w[3] = {0.f, 0.f, 0.f};
#pragma unroll
    for (int r = 0; r < 3; ++r) {
        float m = fmaxf(p[r][0], fmaxf(p[r][1], p[r][2]));
        float e0 = expf(p[r][0] - m), e1 = expf(p[r][1] - m), e2 = expf(p[r][2] - m);
        float inv = 1.0f / (e0 + e1 + e2);
        w[0] += e0 * inv; w[1] += e1 * inv; w[2] += e2 * inv;
    }
    const float third = 1.0f / 3.0f;
    float o0 = third * (w[0] * vr[0][0] + w[1] * vr[1][0] + w[2] * vr[2][0]);
    float o1 = third * (w[0] * vr[0][1] + w[1] * vr[1][1] + w[2] * vr[2][1]);
    out[(size_t)n * HID + head * DHD + d0]     = o0;
    out[(size_t)n * HID + head * DHD + d0 + 1] = o1;
}

__global__ void logits_kernel(const float* __restrict__ h, const float* __restrict__ W,
                              const float* __restrict__ b, float* __restrict__ out) {
    int i = blockIdx.x * blockDim.x + threadIdx.x;
    if (i >= N_NODES * NCLS) return;
    int n = i / NCLS, c = i % NCLS;
    const float* hr = h + (size_t)n * HID;
    float acc = b[c];
    for (int f = 0; f < HID; ++f) acc += hr[f] * W[f * NCLS + c];
    out[i] = acc;
}

// ---------------------------------------------------------------------------
extern "C" void kernel_launch(void* const* d_in, const int* in_sizes, int n_in,
                              void* d_out, int out_size, void* d_ws, size_t ws_size,
                              hipStream_t stream) {
    (void)in_sizes; (void)n_in; (void)out_size; (void)ws_size;
    const float* feature = (const float*)d_in[0];
    const int*   src     = (const int*)d_in[1];
    const int*   dst     = (const int*)d_in[2];
    const float* Wc   = (const float*)d_in[3];
    const float* bc   = (const float*)d_in[4];
    const float* gc   = (const float*)d_in[5];
    const float* bec  = (const float*)d_in[6];
    const float* Wg   = (const float*)d_in[7];
    const float* bg   = (const float*)d_in[8];
    const float* al   = (const float*)d_in[9];
    const float* ar   = (const float*)d_in[10];
    const float* gg   = (const float*)d_in[11];
    const float* beg  = (const float*)d_in[12];
    const float* Ws   = (const float*)d_in[13];
    const float* bs   = (const float*)d_in[14];
    const float* gssc = (const float*)d_in[15];
    const float* bes  = (const float*)d_in[16];
    const float* Wq   = (const float*)d_in[17];
    const float* bq   = (const float*)d_in[18];
    const float* Wk   = (const float*)d_in[19];
    const float* bk   = (const float*)d_in[20];
    const float* Wv   = (const float*)d_in[21];
    const float* bv   = (const float*)d_in[22];
    const float* Wout = (const float*)d_in[23];
    const float* bout = (const float*)d_in[24];

    char* p = (char*)d_ws;
    auto alloc = [&](size_t bytes) -> char* {
        char* r = p;
        p += (bytes + 255) & ~(size_t)255;
        return r;
    };
    const size_t HH  = (size_t)HID * HID;
    const size_t NH  = (size_t)N_NODES * HID;
    const size_t NRH = (size_t)N_NODES * R_REL * HID;

    unsigned short* WcB = (unsigned short*)alloc(L_LAYERS * R_REL * HH * 2);
    unsigned short* WgB = (unsigned short*)alloc(L_LAYERS * R_REL * HH * 2);
    unsigned short* WsB = (unsigned short*)alloc(L_LAYERS * R_REL * HH * 2);
    unsigned short* WqB = (unsigned short*)alloc(L_LAYERS * HH * 2);
    unsigned short* WkB = (unsigned short*)alloc(L_LAYERS * HH * 2);
    unsigned short* WvB = (unsigned short*)alloc(L_LAYERS * HH * 2);
    unsigned short* hb  = (unsigned short*)alloc(NH * 2);
    unsigned short* Sb  = (unsigned short*)alloc(NRH * 2);
    float* S     = (float*)alloc(NRH * 4);
    float* tmp1  = (float*)alloc(NH * 4);
    float* tmp2  = (float*)alloc(NH * 4);
    float* hnext = (float*)alloc(NH * 4);
    float* degO  = (float*)alloc(N_NODES * 4);
    float* degI  = (float*)alloc(N_NODES * 4);
    float* el    = (float*)alloc((size_t)N_NODES * HEADS * 4);
    float* er    = (float*)alloc((size_t)N_NODES * HEADS * 4);
    unsigned* menc = (unsigned*)alloc((size_t)N_NODES * HEADS * 4);
    float* ssum  = (float*)alloc((size_t)N_NODES * HEADS * 4);
    float* ebuf  = (float*)alloc((size_t)E_EDGES * HEADS * 4);
    const int CH = 10000;
    float* qb = (float*)alloc((size_t)CH * R_REL * HID * 4);
    float* kb = (float*)alloc((size_t)CH * R_REL * HID * 4);
    float* vb = (float*)alloc((size_t)CH * R_REL * HID * 4);

    auto cvt = [&](const float* s, unsigned short* d, size_t n) {
        f32_to_bf16_kernel<<<(unsigned)((n + 255) / 256), 256, 0, stream>>>(s, d, n);
    };
    auto gemm = [&](const unsigned short* A, const unsigned short* B,
                    const float* bias, float* C, int M) {
        dim3 g((M + 255) / 256, HID / 64);
        gemm_bf16_wmma<<<g, 256, 0, stream>>>(A, B, bias, C, M);
    };

    // one-time (per call) bf16 weight/feature conversion
    cvt(feature, hb, NH);
    cvt(Wc, WcB, L_LAYERS * R_REL * HH);
    cvt(Wg, WgB, L_LAYERS * R_REL * HH);
    cvt(Ws, WsB, L_LAYERS * R_REL * HH);
    cvt(Wq, WqB, L_LAYERS * HH);
    cvt(Wk, WkB, L_LAYERS * HH);
    cvt(Wv, WvB, L_LAYERS * HH);

    for (int l = 0; l < L_LAYERS; ++l) {
        for (int r = 0; r < R_REL; ++r) {
            const int* sr = src + (size_t)r * E_EDGES;
            const int* dr = dst + (size_t)r * E_EDGES;
            size_t wofs = ((size_t)l * R_REL + r) * HH;
            size_t vofs = ((size_t)l * R_REL + r) * HID;

            // ---- GraphConv path ----
            hipMemsetAsync(degO, 0, (size_t)N_NODES * 4, stream);
            hipMemsetAsync(degI, 0, (size_t)N_NODES * 4, stream);
            degree_kernel<<<(E_EDGES + 255) / 256, 256, 0, stream>>>(sr, dr, degO, degI);
            gemm(hb, WcB + wofs, nullptr, tmp1, N_NODES);
            rowscale_kernel<<<(unsigned)((NH + 255) / 256), 256, 0, stream>>>(tmp1, degO);
            hipMemsetAsync(tmp2, 0, NH * 4, stream);
            scatter_sum_kernel<<<E_EDGES, 128, 0, stream>>>(tmp1, sr, dr, tmp2);
            ln_elu_kernel<<<N_NODES, 256, 0, stream>>>(tmp2, degI, bc + vofs,
                                                       gc + vofs, bec + vofs, S, r, 0);

            // ---- GAT path ----
            gemm(hb, WgB + wofs, nullptr, tmp1, N_NODES);
            gat_elr_kernel<<<(N_NODES * HEADS + 255) / 256, 256, 0, stream>>>(
                tmp1, al + vofs, ar + vofs, el, er);
            hipMemsetAsync(menc, 0, (size_t)N_NODES * HEADS * 4, stream);
            edge_logit_kernel<<<(E_EDGES * HEADS + 255) / 256, 256, 0, stream>>>(
                sr, dr, el, er, ebuf, menc);
            hipMemsetAsync(ssum, 0, (size_t)N_NODES * HEADS * 4, stream);
            edge_expsum_kernel<<<(E_EDGES * HEADS + 255) / 256, 256, 0, stream>>>(
                dr, ebuf, menc, ssum);
            hipMemsetAsync(tmp2, 0, NH * 4, stream);
            gat_scatter_kernel<<<E_EDGES, 128, 0, stream>>>(tmp1, sr, dr, ebuf, ssum, tmp2);
            ln_elu_kernel<<<N_NODES, 256, 0, stream>>>(tmp2, nullptr, bg + vofs,
                                                       gg + vofs, beg + vofs, S, r, 1);

            // ---- skip path ----
            gemm(hb, WsB + wofs, nullptr, tmp1, N_NODES);
            ln_elu_kernel<<<N_NODES, 256, 0, stream>>>(tmp1, nullptr, bs + vofs,
                                                       gssc + vofs, bes + vofs, S, r, 1);
        }

        // ---- relation MHA (chunked q/k/v GEMMs + per-node attention) ----
        cvt(S, Sb, NRH);
        size_t lw = (size_t)l * HH, lb = (size_t)l * HID;
        for (int c0 = 0; c0 < N_NODES; c0 += CH) {
            int nc = (N_NODES - c0) < CH ? (N_NODES - c0) : CH;
            const unsigned short* Sc = Sb + (size_t)c0 * R_REL * HID;
            gemm(Sc, WqB + lw, bq + lb, qb, nc * R_REL);
            gemm(Sc, WkB + lw, bk + lb, kb, nc * R_REL);
            gemm(Sc, WvB + lw, bv + lb, vb, nc * R_REL);
            mha_kernel<<<nc, 256, 0, stream>>>(qb, kb, vb, hnext + (size_t)c0 * HID);
        }
        cvt(hnext, hb, NH);   // input for next layer
    }

    logits_kernel<<<(N_NODES * NCLS + 255) / 256, 256, 0, stream>>>(
        hnext, Wout, bout, (float*)d_out);
}
